// InvResBlock_49409303773284
// MI455X (gfx1250) — matrix operand order repaired and checked by reading, not compiled, not run on hardware
//
#include <hip/hip_runtime.h>

#define BROWS 262144
#define CDIM  64
#define FDIM  256
#define NIT   8

typedef __bf16 bf16;
typedef __attribute__((ext_vector_type(16))) __bf16 v16bf;
typedef __attribute__((ext_vector_type(8)))  __bf16 v8bf;
typedef __attribute__((ext_vector_type(4)))  __bf16 v4bf;
typedef __attribute__((ext_vector_type(8)))  float  v8f;

// LDS layout (bf16 elements). Row strides chosen so stride*2B is a multiple of
// 16B (aligned b128) and per-row bank step keeps 16-lane fragment reads
// conflict-free (step 4 for 264/72, step 20 for 40).
#define SW2 264                      // W2^T rows n=256, k=256 (backward)
#define SW3 72                       // W3^T rows n=256, k=64  (backward)
#define SW1 264                      // W1^T rows n=64,  k=256 (backward)
#define SST 264                      // per-wave stage/d1/d2, 16 rows
#define SW1F 72                      // forward W1 [n=256][k=64]
#define SWCH 40                      // forward W2 chunk [n=256][k=32]
#define SW3F 264                     // forward W3 [n=64][k=256]

#define OFF_WB2 0
#define OFF_WB3 (OFF_WB2 + 256*SW2)
#define OFF_WB1 (OFF_WB3 + 256*SW3)
#define OFF_STG (OFF_WB1 + 64*SW1)
#define OFF_D1  (OFF_STG + 4*16*SST)
#define OFF_D2  (OFF_D1  + 4*16*SST)
#define SMEM_ELEMS (OFF_D2 + 4*16*SST)
#define SMEM_BYTES (SMEM_ELEMS * 2)  // 307200 B <= 320KB WGP LDS

union FragU { v16bf v; v8bf h[2]; };

__device__ __forceinline__ v8f wmma_bf16(v16bf a, v16bf b, v8f c) {
  // (neg_a, A, neg_b, B, c_mod, C, reuse_a, reuse_b)
  return __builtin_amdgcn_wmma_f32_16x16x32_bf16(false, a, false, b,
                                                 (short)0, c, false, false);
}

// A fragment: 16x32 bf16, lane ln = row M; half selects K sub-chunks
// (K = kb..kb+7 & kb+16..kb+23 for half 0; +8 for half 1). Two ds_load_b128.
template <int KS>
__device__ __forceinline__ void load_afr(const bf16* stg, int ln, int half,
                                         v16bf* A) {
#pragma unroll
  for (int ks = 0; ks < KS; ++ks) {
    const bf16* p = stg + ln * SST + ks * 32 + half * 8;
    FragU f;
    f.h[0] = *(const v8bf*)p;
    f.h[1] = *(const v8bf*)(p + 16);
    A[ks] = f.v;
  }
}

// B fragment: 32x16 bf16, lane ln = column N; half selects K sub-chunks
// (K = 0..7 & 16..23 for half 0; +8 for half 1), matching load_afr.
__device__ __forceinline__ v16bf load_bfr(const bf16* Bm, int bs, int col,
                                          int kbase, int half) {
  const bf16* p = Bm + col * bs + kbase + half * 8;
  FragU f;
  f.h[0] = *(const v8bf*)p;
  f.h[1] = *(const v8bf*)(p + 16);
  return f.v;
}

template <int NT, int KS>
__device__ __forceinline__ void gemm_lds(const v16bf* A, const bf16* Bm,
                                         int bs, int ln, int half, v8f* acc) {
#pragma unroll
  for (int ks = 0; ks < KS; ++ks) {
#pragma unroll
    for (int j = 0; j < NT; ++j) {
      acc[j] = wmma_bf16(A[ks], load_bfr(Bm, bs, j * 16 + ln, ks * 32, half),
                         acc[j]);
    }
  }
}

template <int NT>
__device__ __forceinline__ void zero_acc(v8f* acc) {
#pragma unroll
  for (int j = 0; j < NT; ++j)
#pragma unroll
    for (int i = 0; i < 8; ++i) acc[j][i] = 0.f;
}

// D-tile epilogue: z = acc + bias; store elu'(z) -> dbuf, elu(z) -> stg (bf16).
template <int NT>
__device__ __forceinline__ void epi_fwd(const v8f* acc, const float* bias,
                                        bf16* dbuf, bf16* stg, int ln, int half) {
#pragma unroll
  for (int j = 0; j < NT; ++j) {
    float bb = bias[j * 16 + ln];
#pragma unroll
    for (int r = 0; r < 8; ++r) {
      int m = r + 8 * half, n = j * 16 + ln;
      float z = acc[j][r] + bb;
      float e = __expf(z);
      bool pos = z > 0.f;
      dbuf[m * SST + n] = (bf16)(pos ? 1.f : e);
      stg[m * SST + n]  = (bf16)(pos ? z : e - 1.f);
    }
  }
}

// Backward epilogue: u *= elu' mask; stage as next A operand.
template <int NT>
__device__ __forceinline__ void epi_bwd(const v8f* acc, const bf16* dbuf,
                                        bf16* stg, int ln, int half) {
#pragma unroll
  for (int j = 0; j < NT; ++j) {
#pragma unroll
    for (int r = 0; r < 8; ++r) {
      int m = r + 8 * half, n = j * 16 + ln;
      float u = acc[j][r] * (float)dbuf[m * SST + n];
      stg[m * SST + n] = (bf16)u;
    }
  }
}

__device__ __forceinline__ v4bf cvt4(float4 f) {
  v4bf b;
  b[0] = (bf16)f.x; b[1] = (bf16)f.y; b[2] = (bf16)f.z; b[3] = (bf16)f.w;
  return b;
}

// Cooperative forward-W2 chunk stage: dst[n][kk] = W2[n][kb+kk], kk in [0,32).
__device__ __forceinline__ void coop_w2chunk(bf16* dst, const float* W2,
                                             int kb, int tid) {
#pragma unroll 4
  for (int s4 = tid; s4 < 256 * 32 / 4; s4 += 128) {
    int n = s4 >> 3, kk = (s4 & 7) * 4;
    float4 f = *(const float4*)(W2 + n * 256 + kb + kk);
    *(v4bf*)(dst + n * SWCH + kk) = cvt4(f);
  }
}

extern "C" __global__ void __launch_bounds__(128, 1)
invres_wmma(const float* __restrict__ y,  const float* __restrict__ ldj,
            const float* __restrict__ v,  const float* __restrict__ W1,
            const float* __restrict__ b1, const float* __restrict__ W2,
            const float* __restrict__ b2, const float* __restrict__ W3,
            const float* __restrict__ b3, float* __restrict__ zout,
            float* __restrict__ ldout) {
  extern __shared__ bf16 smem[];
  bf16* Wb2 = smem + OFF_WB2;  // [n=256][k=256] = W2[k][n]
  bf16* Wb3 = smem + OFF_WB3;  // [n=256][k=64]  = W3[k][n]
  bf16* Wb1 = smem + OFF_WB1;  // [n=64][k=256]  = W1[k][n]

  const int tid = threadIdx.x;
  const int wave = tid >> 5, lane = tid & 31;
  const int half = lane >> 4, ln = lane & 15;
  bf16* stg = smem + OFF_STG + wave * 16 * SST;
  bf16* d1  = smem + OFF_D1  + wave * 16 * SST;
  bf16* d2  = smem + OFF_D2  + wave * 16 * SST;

  // One-time transposed backward-weight staging (float4 coalesced reads,
  // 16-bit transposed LDS scatter).
  for (int s4 = tid; s4 < 256 * 256 / 4; s4 += 128) {
    int k = s4 >> 6, n = (s4 & 63) * 4;
    float4 f = *(const float4*)(W2 + k * 256 + n);
    Wb2[(n + 0) * SW2 + k] = (bf16)f.x;
    Wb2[(n + 1) * SW2 + k] = (bf16)f.y;
    Wb2[(n + 2) * SW2 + k] = (bf16)f.z;
    Wb2[(n + 3) * SW2 + k] = (bf16)f.w;
  }
  for (int s4 = tid; s4 < 64 * 256 / 4; s4 += 128) {
    int k = s4 >> 6, n = (s4 & 63) * 4;
    float4 f = *(const float4*)(W3 + k * 256 + n);
    Wb3[(n + 0) * SW3 + k] = (bf16)f.x;
    Wb3[(n + 1) * SW3 + k] = (bf16)f.y;
    Wb3[(n + 2) * SW3 + k] = (bf16)f.z;
    Wb3[(n + 3) * SW3 + k] = (bf16)f.w;
  }
  for (int s4 = tid; s4 < 256 * 64 / 4; s4 += 128) {
    int k = s4 >> 4, n = (s4 & 15) * 4;
    float4 f = *(const float4*)(W1 + k * 64 + n);
    Wb1[(n + 0) * SW1 + k] = (bf16)f.x;
    Wb1[(n + 1) * SW1 + k] = (bf16)f.y;
    Wb1[(n + 2) * SW1 + k] = (bf16)f.z;
    Wb1[(n + 3) * SW1 + k] = (bf16)f.w;
  }

  const int rowBase = blockIdx.x * 64 + wave * 16;

  // h0 = elu(y) -> per-wave staging (row-major 16x64)
#pragma unroll
  for (int e = 0; e < 32; ++e) {
    int lin = e * 32 + lane;
    int m = lin >> 6, kk = lin & 63;
    float x = y[(rowBase + m) * CDIM + kk];
    stg[m * SST + kk] = (bf16)(x > 0.f ? x : __expf(x) - 1.f);
  }

  v16bf A[8];
  v8f acc16[16];

  // ---- forward stage 1: z1 = h0 @ W1^T + b1
  // Stage full forward-form W1 (bf16 [n=256][k=64..72]) into the dead d1+d2
  // region (36.9KB <= 67.6KB).
  load_afr<2>(stg, ln, half, A);
  {
    bf16* w1f = smem + OFF_D1;
#pragma unroll 4
    for (int s4 = tid; s4 < 256 * 64 / 4; s4 += 128) {
      int n = s4 >> 4, kk = (s4 & 15) * 4;
      float4 f = *(const float4*)(W1 + n * 64 + kk);
      *(v4bf*)(w1f + n * SW1F + kk) = cvt4(f);
    }
    __syncthreads();  // backward weights + w1f staged
    zero_acc<16>(acc16);
    gemm_lds<16, 2>(A, w1f, SW1F, ln, half, acc16);
  }
  __syncthreads();  // all waves done reading w1f before d1 is written
  epi_fwd<16>(acc16, b1, d1, stg, ln, half);

  // ---- forward stage 2: z2 = h1 @ W2^T + b2
  // Double-buffered K=32 chunks of forward-form W2 through the dead stg (bufA)
  // and d2 (bufB) regions; chunk ks+1 loads overlap chunk ks WMMAs.
  load_afr<8>(stg, ln, half, A);
  zero_acc<16>(acc16);
  {
    bf16* bufA = smem + OFF_STG;
    bf16* bufB = smem + OFF_D2;
    __syncthreads();  // everyone has h1 in registers; stg region now free
    coop_w2chunk(bufA, W2, 0, tid);
    __syncthreads();
#pragma unroll
    for (int ks = 0; ks < 8; ++ks) {
      const bf16* cur = (ks & 1) ? bufB : bufA;
      if (ks < 7) coop_w2chunk((ks & 1) ? bufA : bufB, W2, (ks + 1) * 32, tid);
#pragma unroll
      for (int j = 0; j < 16; ++j) {
        acc16[j] =
            wmma_bf16(A[ks], load_bfr(cur, SWCH, j * 16 + ln, 0, half), acc16[j]);
      }
      __syncthreads();
    }
  }
  epi_fwd<16>(acc16, b2, d2, stg, ln, half);

  // ---- forward stage 3: g = h2 @ W3^T + b3 ; z = y + g
  // Stage forward-form W3 (bf16 [n=64][k=256..264] = 33792B) into the dead
  // stg region (exactly 33792B).
  load_afr<8>(stg, ln, half, A);
  v8f acc4[4];
  {
    bf16* w3f = smem + OFF_STG;
    __syncthreads();  // all waves have h2 in registers
#pragma unroll 4
    for (int s4 = tid; s4 < 64 * 256 / 4; s4 += 128) {
      int n = s4 >> 6, kk = (s4 & 63) * 4;
      float4 f = *(const float4*)(W3 + n * 256 + kk);
      *(v4bf*)(w3f + n * SW3F + kk) = cvt4(f);
    }
    __syncthreads();
    zero_acc<4>(acc4);
    gemm_lds<4, 8>(A, w3f, SW3F, ln, half, acc4);
  }
  __syncthreads();  // all waves done reading w3f before stg gets v0

  float d0D[4][8], v0D[4][8];
#pragma unroll
  for (int t = 0; t < 4; ++t) {
    float bb = b3[t * 16 + ln];
#pragma unroll
    for (int r = 0; r < 8; ++r) {
      int m = r + 8 * half;
      int gi = (rowBase + m) * CDIM + t * 16 + ln;
      float yy = y[gi];
      zout[gi] = yy + acc4[t][r] + bb;
      d0D[t][r] = yy > 0.f ? 1.f : __expf(yy);
      float vv = v[gi];
      v0D[t][r] = vv;
      stg[m * SST + t * 16 + ln] = (bf16)vv;  // w0 = v0 staged as A operand
    }
  }

  float ldacc[8];
#pragma unroll
  for (int r = 0; r < 8; ++r) ldacc[r] = 0.f;

  for (int k = 1; k <= NIT; ++k) {
    // u2 = w @ W3 ; *= d2
    load_afr<2>(stg, ln, half, A);
    zero_acc<16>(acc16);
    gemm_lds<16, 2>(A, Wb3, SW3, ln, half, acc16);
    epi_bwd<16>(acc16, d2, stg, ln, half);
    // u1 = u2d @ W2 ; *= d1
    load_afr<8>(stg, ln, half, A);
    zero_acc<16>(acc16);
    gemm_lds<16, 8>(A, Wb2, SW2, ln, half, acc16);
    epi_bwd<16>(acc16, d1, stg, ln, half);
    // u0 = u1d @ W1 ; w = u0 * d0 ; inner = <w, v0> ; stage w
    load_afr<8>(stg, ln, half, A);
    zero_acc<4>(acc4);
    gemm_lds<4, 8>(A, Wb1, SW1, ln, half, acc4);

    float part[8];
#pragma unroll
    for (int r = 0; r < 8; ++r) part[r] = 0.f;
#pragma unroll
    for (int t = 0; t < 4; ++t) {
#pragma unroll
      for (int r = 0; r < 8; ++r) {
        float w = acc4[t][r] * d0D[t][r];
        part[r] += w * v0D[t][r];
        stg[(r + 8 * half) * SST + t * 16 + ln] = (bf16)w;
      }
    }
    float coef = ((k & 1) ? 1.f : -1.f) / (float)k;
#pragma unroll
    for (int r = 0; r < 8; ++r) {  // reduce across the 16-lane N group
      float s = part[r];
      s += __shfl_xor(s, 1, 32);
      s += __shfl_xor(s, 2, 32);
      s += __shfl_xor(s, 4, 32);
      s += __shfl_xor(s, 8, 32);
      ldacc[r] += coef * s;
    }
  }

  if (ln == 0) {  // lanes 0 (rows m..m+7) and 16 (rows m+8..m+15)
#pragma unroll
    for (int r = 0; r < 8; ++r) {
      int m = rowBase + 8 * half + r;
      ldout[m] = ldj[m] + ldacc[r];
    }
  }
}

extern "C" void kernel_launch(void* const* d_in, const int* in_sizes, int n_in,
                              void* d_out, int out_size, void* d_ws,
                              size_t ws_size, hipStream_t stream) {
  const float* y   = (const float*)d_in[0];
  const float* ldj = (const float*)d_in[1];
  const float* v   = (const float*)d_in[2];
  const float* W1  = (const float*)d_in[3];
  const float* b1  = (const float*)d_in[4];
  const float* W2  = (const float*)d_in[5];
  const float* b2  = (const float*)d_in[6];
  const float* W3  = (const float*)d_in[7];
  const float* b3  = (const float*)d_in[8];
  float* zout  = (float*)d_out;
  float* ldout = zout + (size_t)BROWS * CDIM;

  (void)hipFuncSetAttribute((const void*)invres_wmma,
                            hipFuncAttributeMaxDynamicSharedMemorySize,
                            SMEM_BYTES);
  dim3 grid(BROWS / 64), block(128);
  invres_wmma<<<grid, block, SMEM_BYTES, stream>>>(y, ldj, v, W1, b1, W2, b2,
                                                   W3, b3, zout, ldout);
}
